// SelfAttention3d_85066122264954
// MI455X (gfx1250) — matrix operand order repaired and checked by reading, not compiled, not run on hardware
//
#include <hip/hip_runtime.h>

// ---------------------------------------------------------------------------
// SelfAttention3d for MI455X (gfx1250, wave32, WMMA bf16 16x16x32)
//   B=4, C=128, HEADS=4, ch=32, D=16, H=W=64
//   All WMMA fragments stored pre-swizzled so every operand load is a
//   contiguous 32B LDS read (v16bf). Attention tiles staged via
//   global_load_async_to_lds_b128 (ASYNCcnt) from packed global layout.
//   float->bf16 uses the native hardware convert (fptrunc), not bit tricks.
// ---------------------------------------------------------------------------

typedef __attribute__((ext_vector_type(16))) __bf16 v16bf;
typedef __attribute__((ext_vector_type(8)))  float  v8f;

#define CC   128          // channels
#define NN   65536        // D*H*W per batch
#define BB   4            // batch
#define BN   128          // GEMM N tile
#define WSZ  64           // attention tile side
#define NGRP 8192         // B * C * D attention groups

static __device__ __forceinline__ __bf16 f2bf(float f) {
    return (__bf16)f;     // native v_cvt bf16 path on gfx1250
}

// ---- packed fragment index helpers (ISA 7.12.2 layouts) -------------------
// A fragment 16x32 (MxK): lane = m + 16*h, h = (klocal>>3)&1,
//                         elem = (klocal&7) + 8*(klocal>>4)
static __device__ __forceinline__ int pack_a_idx(int sub, int kc, int m, int klocal) {
    int h = (klocal >> 3) & 1;
    int lp = m + 16 * h;
    int e  = (klocal & 7) + 8 * (klocal >> 4);
    return ((kc * 4 + sub) * 32 + lp) * 16 + e;
}
// B fragment 32x16 (KxN): lane = n + 16*((klocal>>3)&1), elem = (klocal&7) + 8*(klocal>>4)
static __device__ __forceinline__ int pack_b_idx(int sub, int kc, int n, int klocal) {
    int lp = n + 16 * ((klocal >> 3) & 1);
    int e  = (klocal & 7) + 8 * (klocal >> 4);
    return ((kc * 4 + sub) * 32 + lp) * 16 + e;
}

static __device__ __forceinline__ v8f wmma_bf16(v16bf a, v16bf b, v8f c) {
    return __builtin_amdgcn_wmma_f32_16x16x32_bf16(
        false, a, false, b, (short)0, c, false, false);
}

// ---------------------------------------------------------------------------
// Weight conversion: 6 matrices of 128x128 f32 -> bf16, packed consecutively.
// ---------------------------------------------------------------------------
__global__ void cvt_weights_kernel(const float* w0, const float* w1, const float* w2,
                                   const float* w3, const float* w4, const float* w5,
                                   __bf16* out) {
    int i = blockIdx.x * blockDim.x + threadIdx.x;   // < 6*16384
    int m = i >> 14, off = i & 16383;
    const float* src = (m == 0) ? w0 : (m == 1) ? w1 : (m == 2) ? w2
                     : (m == 3) ? w3 : (m == 4) ? w4 : w5;
    out[i] = f2bf(src[off]);
}

// ---------------------------------------------------------------------------
// conv1x1 GEMM:  out[b,o,n] = act( sum_c W[o,c] * X[b,c,n] + bias[o] )
//   256 threads / 8 waves. Workgroup tile: 128(M) x 128(N). K=128 in 4x32.
//   OUT_MODE: 0 = natural bf16, 1 = natural f32, 2 = packed-A (q tiles),
//             3 = packed-B (k/v tiles)
// ---------------------------------------------------------------------------
template <bool IN_F32, bool RELU, int OUT_MODE>
__global__ void conv1x1_kernel(const void* __restrict__ Xin,
                               const __bf16* __restrict__ Wb,
                               const float* __restrict__ bias,
                               void* __restrict__ Out) {
    __shared__ v16bf Wv[CC * CC / 16];   // 32 KB, fragment-packed weights
    __shared__ v16bf Xv[32 * BN / 16];   // 8 KB,  fragment-packed X chunk

    const int t    = threadIdx.x;
    const int wave = t >> 5;
    const int lane = t & 31;

    const int blk = blockIdx.x;
    const int b   = blk / (NN / BN);
    const int n0  = (blk % (NN / BN)) * BN;

    // ---- stage weights, fully vectorized: 8 consecutive bf16 along c land
    //      in one packed 16B octet. idx16 = (mtile*4+kc)*32+lane ; half = kl>>4
    {
        const uint4* ws = (const uint4*)Wb;        // 2048 x 16B
        uint4* wd = (uint4*)Wv;
#pragma unroll
        for (int j = 0; j < 8; ++j) {
            int f = t + j * 256;
            int o = f >> 4;                        // i = f*8 ; o = i>>7
            int c = (f & 15) << 3;                 // multiple of 8
            int mtile = o >> 4, m = o & 15;
            int kc = c >> 5, kl = c & 31;
            int h  = (kl >> 3) & 1;
            int lp = m + 16 * h;
            wd[(((mtile * 4 + kc) * 32 + lp) << 1) + (kl >> 4)] = ws[f];
        }
    }

    v8f acc[8];
#pragma unroll
    for (int j = 0; j < 8; ++j)
#pragma unroll
        for (int r = 0; r < 8; ++r) acc[j][r] = 0.0f;

    __bf16* Xs = (__bf16*)Xv;

    for (int k0 = 0; k0 < CC; k0 += 32) {
        __syncthreads();
        // ---- load 32x128 chunk of X (coalesced), scatter-pack into Xv
        if (IN_F32) {
            const float* X = (const float*)Xin + (size_t)b * CC * NN + n0;
#pragma unroll
            for (int j = 0; j < 4; ++j) {
                int f = t + j * 256;               // float4 index
                int r = f >> 5;                    // i = f*4 ; r = i>>7
                int c0 = (f & 31) << 2;
                float4 d = ((const float4*)(&X[(size_t)(k0 + r) * NN]))[c0 >> 2];
                Xs[pack_b_idx((c0 + 0) >> 4, 0, (c0 + 0) & 15, r)] = f2bf(d.x);
                Xs[pack_b_idx((c0 + 1) >> 4, 0, (c0 + 1) & 15, r)] = f2bf(d.y);
                Xs[pack_b_idx((c0 + 2) >> 4, 0, (c0 + 2) & 15, r)] = f2bf(d.z);
                Xs[pack_b_idx((c0 + 3) >> 4, 0, (c0 + 3) & 15, r)] = f2bf(d.w);
            }
            if (k0 + 32 < CC)                      // global_prefetch_b8
                __builtin_prefetch(&X[(size_t)(k0 + 32 + (t & 31)) * NN + ((t >> 5) << 4)], 0, 1);
        } else {
            const __bf16* X = (const __bf16*)Xin + (size_t)b * CC * NN + n0;
#pragma unroll
            for (int j = 0; j < 2; ++j) {
                int f = t + j * 256;               // uint4 index (8 bf16)
                int r = f >> 4;                    // i = f*8 ; r = i>>7
                int c0 = (f & 15) << 3;
                union { uint4 u; __bf16 h[8]; } d;
                d.u = ((const uint4*)(&X[(size_t)(k0 + r) * NN]))[c0 >> 3];
#pragma unroll
                for (int e = 0; e < 8; ++e)
                    Xs[pack_b_idx((c0 + e) >> 4, 0, (c0 + e) & 15, r)] = d.h[e];
            }
            if (k0 + 32 < CC)
                __builtin_prefetch(&X[(size_t)(k0 + 32 + (t & 31)) * NN + ((t >> 5) << 4)], 0, 1);
        }
        __syncthreads();

        // ---- 8 WMMAs per wave per chunk; all operands contiguous 32B reads
        v16bf a = Wv[(wave * 4 + (k0 >> 5)) * 32 + lane];
#pragma unroll
        for (int jt = 0; jt < 8; ++jt)
            acc[jt] = wmma_bf16(a, Xv[jt * 32 + lane], acc[jt]);
    }

    // ---- epilogue: bias (+relu), store per OUT_MODE
    const int nlane = lane & 15;
    const int mh    = (lane >> 4) << 3;
    float bv[8];
#pragma unroll
    for (int r = 0; r < 8; ++r) bv[r] = bias[wave * 16 + mh + r];

#pragma unroll
    for (int jt = 0; jt < 8; ++jt) {
#pragma unroll
        for (int r = 0; r < 8; ++r) {
            int o = wave * 16 + mh + r;
            float val = acc[jt][r] + bv[r];
            if (RELU) val = val > 0.0f ? val : 0.0f;
            int ncol = n0 + jt * 16 + nlane;
            if (OUT_MODE == 0) {
                ((__bf16*)Out)[(size_t)(b * CC + o) * NN + ncol] = f2bf(val);
            } else if (OUT_MODE == 1) {
                ((float*)Out)[(size_t)(b * CC + o) * NN + ncol] = val;
            } else {
                int d = ncol >> 12, x = (ncol >> 6) & 63, i = ncol & 63;
                size_t gbase = ((size_t)(b * CC + o) * 16 + d) * 4096;
                int kc = x >> 5, kl = x & 31, sub = i >> 4;
                int idx = (OUT_MODE == 2) ? pack_a_idx(sub, kc, i & 15, kl)
                                          : pack_b_idx(sub, kc, i & 15, kl);
                ((__bf16*)Out)[gbase + idx] = f2bf(val);
            }
        }
    }
}

// ---------------------------------------------------------------------------
// Attention over one (b,head,c,d) group: S = Q^T K ; softmax rows ; O = S V
//   128 threads / 4 waves. q/k/v arrive pre-packed (8 KB contiguous/group)
//   and are staged with global_load_async_to_lds_b128 (ASYNCcnt path).
//   Output (natural layout) overwrites the q-tile region in place.
// ---------------------------------------------------------------------------
__global__ void attention_kernel(const __bf16* __restrict__ qp,
                                 const __bf16* __restrict__ kp,
                                 const __bf16* __restrict__ vp,
                                 __bf16* __restrict__ ob) {
    __shared__ v16bf Qv[256];           // 8 KB packed-A fragments of Q^T
    __shared__ v16bf Kv[256];           // 8 KB packed-B fragments of K
    __shared__ v16bf Vv[256];           // 8 KB packed-B fragments of V
    __shared__ v16bf Sbv[256];          // 8 KB packed-A fragments of softmax(S)
    __shared__ float Sf[WSZ * WSZ];     // 16 KB raw scores

    const int t    = threadIdx.x;       // 128 threads
    const int wave = t >> 5;
    const int lane = t & 31;
    const size_t base = (size_t)blockIdx.x * (WSZ * WSZ);

    // ---- async stage: 3 x 8KB, 16B per issue, 12 issues per thread
    {
        unsigned ql = (unsigned)(uintptr_t)&Qv[0];
        unsigned kl = (unsigned)(uintptr_t)&Kv[0];
        unsigned vl = (unsigned)(uintptr_t)&Vv[0];
        const char* qg = (const char*)(qp + base);
        const char* kg = (const char*)(kp + base);
        const char* vg = (const char*)(vp + base);
#pragma unroll
        for (int j = 0; j < 4; ++j) {
            int off = (t + j * 128) * 16;
            asm volatile("global_load_async_to_lds_b128 %0, %1, off"
                         :: "v"(ql + off), "v"(qg + off) : "memory");
            asm volatile("global_load_async_to_lds_b128 %0, %1, off"
                         :: "v"(kl + off), "v"(kg + off) : "memory");
            asm volatile("global_load_async_to_lds_b128 %0, %1, off"
                         :: "v"(vl + off), "v"(vg + off) : "memory");
        }
        asm volatile("s_wait_asynccnt 0x0" ::: "memory");
    }
    __syncthreads();

    // ---- S = Q^T K : wave owns S rows [16w,16w+16), 4 col sub-tiles
    v8f s[4];
#pragma unroll
    for (int j = 0; j < 4; ++j)
#pragma unroll
        for (int r = 0; r < 8; ++r) s[j][r] = 0.0f;

#pragma unroll
    for (int kc = 0; kc < 2; ++kc) {
        v16bf a = Qv[(kc * 4 + wave) * 32 + lane];
#pragma unroll
        for (int jt = 0; jt < 4; ++jt)
            s[jt] = wmma_bf16(a, Kv[(kc * 4 + jt) * 32 + lane], s[jt]);
    }

    {   // spill raw scores (f32) in natural layout
        const int n  = lane & 15;
        const int mh = (lane >> 4) << 3;
#pragma unroll
        for (int jt = 0; jt < 4; ++jt)
#pragma unroll
            for (int r = 0; r < 8; ++r)
                Sf[(wave * 16 + mh + r) * WSZ + jt * 16 + n] = s[jt][r];
    }
    __syncthreads();

    // ---- row softmax; result written directly as packed-A fragments
    if (t < WSZ) {
        float* row = Sf + t * WSZ;
        float mx = row[0];
#pragma unroll
        for (int j = 1; j < WSZ; ++j) mx = row[j] > mx ? row[j] : mx;
        float sum = 0.0f;
#pragma unroll
        for (int j = 0; j < WSZ; ++j) { float ev = __expf(row[j] - mx); row[j] = ev; sum += ev; }
        float inv = 1.0f / sum;
        __bf16* Sbs = (__bf16*)Sbv;
        const int sub = t >> 4, m = t & 15;
#pragma unroll
        for (int j = 0; j < WSZ; ++j)
            Sbs[pack_a_idx(sub, j >> 5, m, j & 31)] = f2bf(row[j] * inv);
    }
    __syncthreads();

    // ---- O = softmax(S) @ V
    v8f o[4];
#pragma unroll
    for (int j = 0; j < 4; ++j)
#pragma unroll
        for (int r = 0; r < 8; ++r) o[j][r] = 0.0f;

#pragma unroll
    for (int kc = 0; kc < 2; ++kc) {
        v16bf a = Sbv[(kc * 4 + wave) * 32 + lane];
#pragma unroll
        for (int jt = 0; jt < 4; ++jt)
            o[jt] = wmma_bf16(a, Vv[(kc * 4 + jt) * 32 + lane], o[jt]);
    }

    {   // store O (bf16, natural layout) over the q-tile region
        const int n  = lane & 15;
        const int mh = (lane >> 4) << 3;
#pragma unroll
        for (int jt = 0; jt < 4; ++jt)
#pragma unroll
            for (int r = 0; r < 8; ++r)
                ob[base + (size_t)(wave * 16 + mh + r) * WSZ + jt * 16 + n] = f2bf(o[jt][r]);
    }
}

// ---------------------------------------------------------------------------
// Host-side orchestration (graph-capture safe: only kernel launches on stream)
// ---------------------------------------------------------------------------
extern "C" void kernel_launch(void* const* d_in, const int* in_sizes, int n_in,
                              void* d_out, int out_size, void* d_ws, size_t ws_size,
                              hipStream_t stream) {
    (void)in_sizes; (void)n_in; (void)out_size; (void)ws_size;

    const float* x   = (const float*)d_in[0];
    const float* Wq  = (const float*)d_in[1];  const float* bq  = (const float*)d_in[2];
    const float* Wk  = (const float*)d_in[3];  const float* bk  = (const float*)d_in[4];
    const float* Wv  = (const float*)d_in[5];  const float* bv  = (const float*)d_in[6];
    const float* Wa1 = (const float*)d_in[7];  const float* ba1 = (const float*)d_in[8];
    const float* Wa2 = (const float*)d_in[9];  const float* ba2 = (const float*)d_in[10];
    const float* Wf  = (const float*)d_in[11]; const float* bf  = (const float*)d_in[12];

    const size_t NVC = (size_t)BB * CC * NN;           // 33,554,432 elements
    __bf16* qb   = (__bf16*)d_ws;                      // 64 MB each
    __bf16* kbuf = qb + NVC;
    __bf16* vbuf = kbuf + NVC;
    __bf16* wbf  = vbuf + NVC;                         // 6 x 16384 bf16 weights

    const dim3 cgrid(BB * (NN / BN));                  // 2048 workgroups
    const dim3 cblk(256);

    // 0) weights f32 -> bf16
    cvt_weights_kernel<<<dim3((6 * 16384) / 256), dim3(256), 0, stream>>>(
        Wq, Wk, Wv, Wa1, Wa2, Wf, wbf);

    // 1) q,k,v convs; outputs pre-packed into WMMA fragment layout per group.
    //    x (134 MB) stays resident in the 192 MB L2 across the three passes.
    conv1x1_kernel<true,  false, 2><<<cgrid, cblk, 0, stream>>>(x, wbf + 0 * 16384, bq, qb);
    conv1x1_kernel<true,  false, 3><<<cgrid, cblk, 0, stream>>>(x, wbf + 1 * 16384, bk, kbuf);
    conv1x1_kernel<true,  false, 3><<<cgrid, cblk, 0, stream>>>(x, wbf + 2 * 16384, bv, vbuf);

    // 2) per-group attention (async-staged tiles); output overwrites q buffer
    attention_kernel<<<dim3(NGRP), dim3(128), 0, stream>>>(qb, kbuf, vbuf, qb);

    // 3) conv chain: relu(Wa1) -> Wa2 -> Wf (final f32 to d_out)
    conv1x1_kernel<false, true,  0><<<cgrid, cblk, 0, stream>>>(qb,   wbf + 3 * 16384, ba1, kbuf);
    conv1x1_kernel<false, false, 0><<<cgrid, cblk, 0, stream>>>(kbuf, wbf + 4 * 16384, ba2, vbuf);
    conv1x1_kernel<false, false, 1><<<cgrid, cblk, 0, stream>>>(vbuf, wbf + 5 * 16384, bf,  d_out);
}
